// GenericAttention_10007273799987
// MI455X (gfx1250) — matrix operand order repaired and checked by reading, not compiled, not run on hardware
//
#include <hip/hip_runtime.h>
#include <hip/hip_bf16.h>

// ---------------- problem constants ----------------
#define BB    2
#define NN    2048
#define FF    1024
#define HH    16
#define DH    64
#define KD    1024
#define SCALE 0.125f   // 1/sqrt(64)

typedef __attribute__((ext_vector_type(16))) __bf16 v16bf;
typedef __attribute__((ext_vector_type(8)))  float  v8f;
typedef int v4i __attribute__((vector_size(16)));   // matches builtin param type

union BFrag { v16bf v; uint4 q[2]; };

__device__ __forceinline__ v8f wmma_bf16(const v16bf a, const v16bf b, v8f c) {
    return __builtin_amdgcn_wmma_f32_16x16x32_bf16(false, a, false, b, (short)0, c,
                                                   false, false);
}

__device__ __forceinline__ unsigned short f2bf(float f) {
    unsigned u = __float_as_uint(f);
    u += 0x7fffu + ((u >> 16) & 1u);        // round-to-nearest-even
    return (unsigned short)(u >> 16);
}

// ---------------- CDNA5 async Global->LDS copy (ASYNCcnt-tracked) ------------
#if __has_builtin(__builtin_amdgcn_global_load_async_to_lds_b128)
#define HAVE_ASYNC_BUILTIN 1
#endif

__device__ __forceinline__ void cp16_async(const unsigned short* g,
                                           unsigned short* l) {
#ifdef HAVE_ASYNC_BUILTIN
    __builtin_amdgcn_global_load_async_to_lds_b128(
        (__attribute__((address_space(1))) v4i*)g,
        (__attribute__((address_space(3))) v4i*)l, 0, 0);
#else
    // VDST = LDS byte address (low 32 bits of generic shared pointer),
    // VADDR = 64-bit global address, no SADDR.
    asm volatile("global_load_async_to_lds_b128 %0, %1, off"
                 :: "v"((unsigned)(unsigned long long)l),
                    "v"((unsigned long long)g)
                 : "memory");
#endif
}

template <int N>
__device__ __forceinline__ void wait_async() {
#if __has_builtin(__builtin_amdgcn_s_wait_asynccnt)
    __builtin_amdgcn_s_wait_asynccnt(N);
#else
    asm volatile("s_wait_asynccnt %0" :: "i"(N) : "memory");
#endif
}

// ---------------- 1) fp32 -> bf16 convert (float4 vectorized) ----------------
__global__ void cvt_f32_bf16(const float* __restrict__ src,
                             unsigned short* __restrict__ dst, int n) {
    int i = (blockIdx.x * blockDim.x + threadIdx.x) * 4;
    if (i < n) {
        float4 f = *(const float4*)(src + i);
        ushort4 o;
        o.x = f2bf(f.x); o.y = f2bf(f.y); o.z = f2bf(f.z); o.w = f2bf(f.w);
        *(ushort4*)(dst + i) = o;
    }
}

// ---------------- 2) weight transpose + convert: Wt[n][k] = bf16(W[k][n]) ----
__global__ void transpose_w(const float* __restrict__ W,
                            unsigned short* __restrict__ Wt, int dim) {
    __shared__ float tile[32][33];
    int x  = blockIdx.x * 32 + threadIdx.x;   // column n of W
    int y0 = blockIdx.y * 32;                 // row block k
    #pragma unroll
    for (int j = threadIdx.y; j < 32; j += 8)
        tile[j][threadIdx.x] = W[(size_t)(y0 + j) * dim + x];
    __syncthreads();
    int k = y0 + threadIdx.x;
    #pragma unroll
    for (int j = threadIdx.y; j < 32; j += 8)
        Wt[(size_t)(blockIdx.x * 32 + j) * dim + k] = f2bf(tile[threadIdx.x][j]);
}

// ---------------- 3) bf16 WMMA GEMM, LDS-staged + async double buffering -----
// Block: 128 threads (4 waves). Tile: 128(M) x 64(N), K-step 32.
// Wave w computes rows w*32..w*32+31 (2 M-frags) x 64 cols (4 N-frags).
// mode 0: out bf16 head-major   Qh/Kh [B][H][N][64]
// mode 2: out bf16 d-major      Vt    [B][H][64][N]
// mode 3: out fp32 flat         [M][1024]  (final output)
__global__ void __launch_bounds__(128)
gemm_bf16_wmma(const unsigned short* __restrict__ A,
               const unsigned short* __restrict__ Bt,
               const float* __restrict__ bias,
               unsigned short* __restrict__ outU,
               float* __restrict__ outF,
               int mode) {
    constexpr int LDR = 40;   // padded LDS row (elements): 80B = 20 banks, 16B-aligned
    __shared__ alignas(16) unsigned short sA[2][128 * LDR];
    __shared__ alignas(16) unsigned short sB[2][64 * LDR];

    const int tid  = threadIdx.x;
    const int lane = tid & 31;
    const int wv   = tid >> 5;
    const int m16  = lane & 15;
    const int hf   = lane >> 4;
    const int rowBlk = blockIdx.y * 128;
    const int colBlk = blockIdx.x * 64;

    // stage one 32-wide K slice of A (128x32) and B (64x32) into buf
    auto issue = [&](int buf, int k0) {
        #pragma unroll
        for (int i = 0; i < 4; ++i) {           // A: 512 chunks of 16B
            int L = i * 128 + tid;
            int row = L >> 2, ch = L & 3;
            cp16_async(A + (size_t)(rowBlk + row) * KD + k0 + ch * 8,
                       &sA[buf][row * LDR + ch * 8]);
        }
        #pragma unroll
        for (int i = 0; i < 2; ++i) {           // B: 256 chunks of 16B
            int L = i * 128 + tid;
            int col = L >> 2, ch = L & 3;
            cp16_async(Bt + (size_t)(colBlk + col) * KD + k0 + ch * 8,
                       &sB[buf][col * LDR + ch * 8]);
        }
    };

    v8f zero = {};
    v8f acc[2][4] = {{zero, zero, zero, zero}, {zero, zero, zero, zero}};

    issue(0, 0);
    for (int kt = 0; kt < KD; kt += 32) {
        const int cur = (kt >> 5) & 1;
        if (kt + 32 < KD) { issue(cur ^ 1, kt + 32); wait_async<6>(); }
        else              { wait_async<0>(); }
        __syncthreads();   // staged slice visible to all waves

        BFrag a[2], b[4];
        #pragma unroll
        for (int t = 0; t < 2; ++t) {
            const unsigned short* ap =
                &sA[cur][(wv * 32 + t * 16 + m16) * LDR + hf * 8];
            a[t].q[0] = *(const uint4*)ap;         // k = 8*hf + e
            a[t].q[1] = *(const uint4*)(ap + 16);  // k = 16 + 8*hf + e
        }
        #pragma unroll
        for (int f = 0; f < 4; ++f) {
            const unsigned short* bp = &sB[cur][(f * 16 + m16) * LDR + hf * 8];
            b[f].q[0] = *(const uint4*)bp;         // k = 8*hf + e
            b[f].q[1] = *(const uint4*)(bp + 16);  // k = 16 + 8*hf + e
            #pragma unroll
            for (int t = 0; t < 2; ++t)
                acc[t][f] = wmma_bf16(a[t].v, b[f].v, acc[t][f]);
        }
        __syncthreads();   // all reads done before buffer is re-staged
    }

    #pragma unroll
    for (int t = 0; t < 2; ++t) {
        #pragma unroll
        for (int f = 0; f < 4; ++f) {
            int cc = colBlk + f * 16 + m16;
            float bv = bias[cc];
            #pragma unroll
            for (int r = 0; r < 8; ++r) {
                int rr  = rowBlk + wv * 32 + t * 16 + r + hf * 8;
                float v = acc[t][f][r] + bv;
                int bb = rr >> 11, n = rr & (NN - 1);
                int h  = cc >> 6,  d = cc & (DH - 1);
                if (mode == 0) {
                    outU[(((size_t)bb * HH + h) * NN + n) * DH + d] = f2bf(v);
                } else if (mode == 2) {
                    outU[(((size_t)bb * HH + h) * DH + d) * NN + n] = f2bf(v);
                } else {
                    outF[(size_t)rr * FF + cc] = v;
                }
            }
        }
    }
}

// ---------------- 4) flash attention over bf16 heads -------------------------
// Qh,Kh: [B*H][N][64] bf16 ; Vt: [B*H][64][N] bf16 ; O: [B][N][F] bf16
__global__ void __launch_bounds__(128)
attn_wmma(const unsigned short* __restrict__ Qh,
          const unsigned short* __restrict__ Kh,
          const unsigned short* __restrict__ Vt,
          unsigned short* __restrict__ O) {
    const int lane = threadIdx.x & 31;
    const int wv   = threadIdx.x >> 5;
    const int m16  = lane & 15;
    const int hf   = lane >> 4;
    const int bh    = blockIdx.y;                 // b*16 + h
    const int qBase = blockIdx.x * 64 + wv * 16;  // this wave's 16 query rows

    __shared__ alignas(16) unsigned short ldsP[4][16][32];  // per-wave P staging

    const unsigned short* qrow = Qh + ((size_t)bh * NN + (qBase + m16)) * DH;
    BFrag aq0, aq1;
    aq0.q[0] = *(const uint4*)(qrow + hf * 8);
    aq0.q[1] = *(const uint4*)(qrow + 16 + hf * 8);
    aq1.q[0] = *(const uint4*)(qrow + 32 + hf * 8);
    aq1.q[1] = *(const uint4*)(qrow + 48 + hf * 8);

    float mi[8], li[8];
    #pragma unroll
    for (int r = 0; r < 8; ++r) { mi[r] = -1e30f; li[r] = 0.0f; }
    v8f zero = {};
    v8f o[4] = {zero, zero, zero, zero};

    for (int kb = 0; kb < NN; kb += 32) {
        // prefetch next K/V tile toward L0/L2 (global_prefetch_b8)
        if (kb + 32 < NN) {
            __builtin_prefetch(Kh + ((size_t)bh * NN + kb + 32 + m16) * DH, 0, 1);
            __builtin_prefetch(Vt + ((size_t)bh * DH + m16) * NN + kb + 32, 0, 1);
        }

        // ---- S = Q @ K^T over a 32-key tile: two 16-col fragments ----
        v8f s0 = zero, s1 = zero;
        #pragma unroll
        for (int c = 0; c < 2; ++c) {
            const unsigned short* kr =
                Kh + ((size_t)bh * NN + kb + c * 16 + m16) * DH + hf * 8;
            BFrag b0, b1;
            b0.q[0] = ((const uint4*)kr)[0];          // d = 8*hf + e
            b0.q[1] = ((const uint4*)(kr + 16))[0];   // d = 16 + 8*hf + e
            b1.q[0] = ((const uint4*)(kr + 32))[0];   // d = 32 + 8*hf + e
            b1.q[1] = ((const uint4*)(kr + 48))[0];
            v8f& s = c ? s1 : s0;
            s = wmma_bf16(aq0.v, b0.v, s);
            s = wmma_bf16(aq1.v, b1.v, s);
        }

        // ---- online softmax (rows live in 16-lane halves; xor<=8 stays inside) ----
        #pragma unroll
        for (int r = 0; r < 8; ++r) {
            float a0 = s0[r] * SCALE;
            float a1 = s1[r] * SCALE;
            float rmax = fmaxf(a0, a1);
            #pragma unroll
            for (int off = 8; off >= 1; off >>= 1)
                rmax = fmaxf(rmax, __shfl_xor(rmax, off, 32));
            float mnew  = fmaxf(mi[r], rmax);
            float alpha = __expf(mi[r] - mnew);
            float p0 = __expf(a0 - mnew);
            float p1 = __expf(a1 - mnew);
            float rsum = p0 + p1;
            #pragma unroll
            for (int off = 8; off >= 1; off >>= 1)
                rsum += __shfl_xor(rsum, off, 32);
            li[r] = li[r] * alpha + rsum;
            mi[r] = mnew;
            #pragma unroll
            for (int f = 0; f < 4; ++f) o[f][r] = o[f][r] * alpha;
            int row = r + hf * 8;                       // C-layout row mapping
            ldsP[wv][row][m16]      = f2bf(p0);
            ldsP[wv][row][16 + m16] = f2bf(p1);
        }
        __syncthreads();

        // re-read P in A-fragment layout (two 16B LDS loads per lane)
        BFrag ap;
        ap.q[0] = *(const uint4*)&ldsP[wv][m16][hf * 8];
        ap.q[1] = *(const uint4*)&ldsP[wv][m16][16 + hf * 8];

        // ---- O += P @ V (contraction over 32 keys, contiguous Vt rows) ----
        #pragma unroll
        for (int f = 0; f < 4; ++f) {
            const unsigned short* vr =
                Vt + ((size_t)bh * DH + f * 16 + m16) * NN + kb + hf * 8;
            BFrag bv;
            bv.q[0] = ((const uint4*)vr)[0];
            bv.q[1] = ((const uint4*)(vr + 16))[0];
            o[f] = wmma_bf16(ap.v, bv.v, o[f]);
        }
        __syncthreads();
    }

    // ---- normalize and write merged-head bf16 output [B][N][F] ----
    const int h  = bh & (HH - 1);
    const int bb = bh >> 4;
    #pragma unroll
    for (int r = 0; r < 8; ++r) {
        float inv = 1.0f / li[r];
        int n = qBase + r + hf * 8;
        #pragma unroll
        for (int f = 0; f < 4; ++f)
            O[((size_t)bb * NN + n) * FF + h * DH + f * 16 + m16] =
                f2bf(o[f][r] * inv);
    }
}

// ---------------- host-side orchestration ------------------------------------
extern "C" void kernel_launch(void* const* d_in, const int* in_sizes, int n_in,
                              void* d_out, int out_size, void* d_ws, size_t ws_size,
                              hipStream_t stream) {
    const float* q  = (const float*)d_in[0];
    const float* k  = (const float*)d_in[1];
    const float* v  = (const float*)d_in[2];
    const float* Wq = (const float*)d_in[3];
    const float* bq = (const float*)d_in[4];
    const float* Wk = (const float*)d_in[5];
    const float* bk = (const float*)d_in[6];
    const float* Wv = (const float*)d_in[7];
    const float* bv = (const float*)d_in[8];
    const float* Wo = (const float*)d_in[9];
    const float* bo = (const float*)d_in[10];
    float* out = (float*)d_out;

    const size_t ACT = (size_t)BB * NN * FF;      // 4,194,304 elements
    const size_t WEL = (size_t)FF * FF;           // 1,048,576 elements
    char* ws = (char*)d_ws;
    size_t off = 0;
    auto carve = [&](size_t bytes) {
        void* p = ws + off;
        off += (bytes + 255) & ~(size_t)255;
        return p;
    };
    unsigned short* qb  = (unsigned short*)carve(ACT * 2);
    unsigned short* kb  = (unsigned short*)carve(ACT * 2);
    unsigned short* vb  = (unsigned short*)carve(ACT * 2);
    unsigned short* wqt = (unsigned short*)carve(WEL * 2);
    unsigned short* wkt = (unsigned short*)carve(WEL * 2);
    unsigned short* wvt = (unsigned short*)carve(WEL * 2);
    unsigned short* wot = (unsigned short*)carve(WEL * 2);
    unsigned short* Qh  = (unsigned short*)carve(ACT * 2);
    unsigned short* Kh  = (unsigned short*)carve(ACT * 2);
    unsigned short* Vt  = (unsigned short*)carve(ACT * 2);
    unsigned short* aO  = (unsigned short*)carve(ACT * 2);
    (void)ws_size; (void)n_in; (void)in_sizes; (void)out_size;

    // 1) activations -> bf16
    int cvtGrid = (int)(ACT / (256 * 4));
    cvt_f32_bf16<<<cvtGrid, 256, 0, stream>>>(q, qb, (int)ACT);
    cvt_f32_bf16<<<cvtGrid, 256, 0, stream>>>(k, kb, (int)ACT);
    cvt_f32_bf16<<<cvtGrid, 256, 0, stream>>>(v, vb, (int)ACT);

    // 2) weights -> transposed bf16
    dim3 tb(32, 8), tg(FF / 32, FF / 32);
    transpose_w<<<tg, tb, 0, stream>>>(Wq, wqt, FF);
    transpose_w<<<tg, tb, 0, stream>>>(Wk, wkt, FF);
    transpose_w<<<tg, tb, 0, stream>>>(Wv, wvt, FF);
    transpose_w<<<tg, tb, 0, stream>>>(Wo, wot, FF);

    // 3) projections (WMMA GEMMs, 128x64 block tiles)
    dim3 gb(128), gg(FF / 64, (BB * NN) / 128);
    gemm_bf16_wmma<<<gg, gb, 0, stream>>>(qb, wqt, bq, Qh, nullptr, 0);
    gemm_bf16_wmma<<<gg, gb, 0, stream>>>(kb, wkt, bk, Kh, nullptr, 0);
    gemm_bf16_wmma<<<gg, gb, 0, stream>>>(vb, wvt, bv, Vt, nullptr, 2);

    // 4) flash attention (WMMA)
    dim3 ab(128), ag(NN / 64, BB * HH);
    attn_wmma<<<ag, ab, 0, stream>>>(Qh, Kh, Vt, aO);

    // 5) output projection -> fp32 d_out
    gemm_bf16_wmma<<<gg, gb, 0, stream>>>(aO, wot, bo, nullptr, out, 3);
}
